// DeepONet_15530601742786
// MI455X (gfx1250) — matrix operand order repaired and checked by reading, not compiled, not run on hardware
//
#include <hip/hip_runtime.h>

// ---------------------------------------------------------------------------
// Problem constants (match reference)
// ---------------------------------------------------------------------------
#define N_NODES  100000
#define N_EDGES  800000
#define N_GRAPHS 64
#define HID      64
#define BN_EPS   1e-5f

typedef __attribute__((ext_vector_type(16))) _Float16 v16h;
typedef __attribute__((ext_vector_type(8)))  float    v8f;

// ---------------------------------------------------------------------------
// Utility kernels
// ---------------------------------------------------------------------------
__global__ void zero_kernel(float* __restrict__ p, int n) {
    int i = blockIdx.x * blockDim.x + threadIdx.x;
    if (i < n) p[i] = 0.0f;
}

// deg[dst] += 1 over all edges (f32 hw atomic; counts < 2^24 so exact)
__global__ void degree_kernel(const int* __restrict__ dst, float* __restrict__ deg, int e) {
    int i = blockIdx.x * blockDim.x + threadIdx.x;
    if (i < e) unsafeAtomicAdd(&deg[dst[i]], 1.0f);
}

// dis[n] = rsqrt(deg[n] + 1)   (in place)
__global__ void rsqrt_kernel(float* __restrict__ deg, int n) {
    int i = blockIdx.x * blockDim.x + threadIdx.x;
    if (i < n) deg[i] = rsqrtf(deg[i] + 1.0f);
}

// ---------------------------------------------------------------------------
// Node-feature GEMM: OUT[N,64] = H[N,64] @ W[64,64]  via v_wmma_f32_16x16x32_f16
// One wave owns a 16x64 tile; B fragments (all of W) built once per wave.
// ---------------------------------------------------------------------------
__global__ void gemm_node_wmma(const float* __restrict__ H, const float* __restrict__ W,
                               float* __restrict__ OUT, int nrows) {
    const int lane  = threadIdx.x & 31;
    const int g     = lane >> 4;        // lane half: 0 or 1
    const int r16   = lane & 15;        // row / col within 16
    const int wid   = (blockIdx.x * blockDim.x + threadIdx.x) >> 5;
    const int nwav  = (gridDim.x * blockDim.x) >> 5;

    // --- B fragments: 4 n-tiles x 2 K-chunks of W (32x16 f16 each) ---------
    // half e of fragment: K = kc*32 + 16*g + e ; col = nt*16 + (lane%16)
    v16h bfrag[4][2];
#pragma unroll
    for (int nt = 0; nt < 4; ++nt) {
#pragma unroll
        for (int kc = 0; kc < 2; ++kc) {
            v16h b;
            const int col = nt * 16 + r16;
#pragma unroll
            for (int e = 0; e < 16; ++e)
                b[e] = (_Float16)W[(kc * 32 + 16 * g + e) * HID + col];
            bfrag[nt][kc] = b;
        }
    }

    const int ntiles = nrows >> 4;      // nrows divisible by 16 (100000/16 = 6250)
    for (int t = wid; t < ntiles; t += nwav) {
        const float* __restrict__ hrow = H + (size_t)(t * 16 + r16) * HID;

        // A fragments (16x32 f16): half e -> K = kc*32 + 16*g + e
        v16h afrag[2];
#pragma unroll
        for (int kc = 0; kc < 2; ++kc) {
            v16h a;
            const int kb = kc * 32 + 16 * g;
#pragma unroll
            for (int e = 0; e < 16; ++e)
                a[e] = (_Float16)hrow[kb + e];
            afrag[kc] = a;
        }

#pragma unroll
        for (int nt = 0; nt < 4; ++nt) {
            v8f acc = {};
            acc = __builtin_amdgcn_wmma_f32_16x16x32_f16(false, afrag[0], false, bfrag[nt][0],
                                                         (short)0, acc, false, false);
            acc = __builtin_amdgcn_wmma_f32_16x16x32_f16(false, afrag[1], false, bfrag[nt][1],
                                                         (short)0, acc, false, false);
            // D layout: VGPR v -> row = v + 8*g ; col = lane%16
#pragma unroll
            for (int v = 0; v < 8; ++v)
                OUT[(size_t)(t * 16 + v + 8 * g) * HID + nt * 16 + r16] = acc[v];
        }
    }
}

// ---------------------------------------------------------------------------
// Edge scatter: agg[dst] += enorm * hw[src]   (one wave per edge, 2 feats/lane)
// ---------------------------------------------------------------------------
__global__ void gcn_scatter(const float* __restrict__ HW, const int* __restrict__ src,
                            const int* __restrict__ dst, const float* __restrict__ dis,
                            float* __restrict__ AGG, int nedges) {
    const int wid  = (blockIdx.x * blockDim.x + threadIdx.x) >> 5;
    const int lane = threadIdx.x & 31;
    if (wid >= nedges) return;
    const int s = src[wid];
    const int d = dst[wid];
    const float en = dis[s] * dis[d];
    const float2 v = ((const float2*)(HW + (size_t)s * HID))[lane];
    float* ap = AGG + (size_t)d * HID + 2 * lane;
    unsafeAtomicAdd(ap,     v.x * en);
    unsafeAtomicAdd(ap + 1, v.y * en);
}

// ---------------------------------------------------------------------------
// Fused self-loop + bias + BatchNorm + ReLU   (agg updated in place -> new h)
// ---------------------------------------------------------------------------
__global__ void gcn_finalize(float* __restrict__ AGG, const float* __restrict__ HW,
                             const float* __restrict__ dis, const float* __restrict__ b,
                             const float* __restrict__ bg, const float* __restrict__ bb,
                             const float* __restrict__ bm, const float* __restrict__ bv,
                             int nnodes) {
    int idx = blockIdx.x * blockDim.x + threadIdx.x;
    if (idx >= nnodes * HID) return;
    const int n = idx >> 6, f = idx & 63;
    const float dn = dis[n];
    float val = AGG[idx] + (dn * dn) * HW[idx] + b[f];
    val = (val - bm[f]) * rsqrtf(bv[f] + BN_EPS) * bg[f] + bb[f];
    AGG[idx] = fmaxf(val, 0.0f);
}

// ---------------------------------------------------------------------------
// Global mean pool (sum stage): LDS accumulation (ds_add_f32), sparse flush
// ---------------------------------------------------------------------------
__global__ void pool_kernel(const float* __restrict__ H, const int* __restrict__ batch,
                            float* __restrict__ pooled, float* __restrict__ cnt, int nnodes) {
    __shared__ float sh[N_GRAPHS * HID];
    __shared__ float shc[N_GRAPHS];
    for (int i = threadIdx.x; i < N_GRAPHS * HID; i += blockDim.x) sh[i] = 0.0f;
    if (threadIdx.x < N_GRAPHS) shc[threadIdx.x] = 0.0f;
    __syncthreads();

    const int lane = threadIdx.x & 31;
    const int wid  = (blockIdx.x * blockDim.x + threadIdx.x) >> 5;
    const int nwav = (gridDim.x * blockDim.x) >> 5;
    for (int n = wid; n < nnodes; n += nwav) {
        const int gph = batch[n];
        const float* hn = H + (size_t)n * HID;
        atomicAdd(&sh[gph * HID + lane],        hn[lane]);
        atomicAdd(&sh[gph * HID + lane + 32],   hn[lane + 32]);
        if (lane == 0) atomicAdd(&shc[gph], 1.0f);
    }
    __syncthreads();

    for (int i = threadIdx.x; i < N_GRAPHS * HID; i += blockDim.x) {
        const float v = sh[i];
        if (v != 0.0f) unsafeAtomicAdd(&pooled[i], v);
    }
    if (threadIdx.x < N_GRAPHS) {
        const float c = shc[threadIdx.x];
        if (c != 0.0f) unsafeAtomicAdd(&cnt[threadIdx.x], c);
    }
}

// pooled mean: pm[g][f] = pooled[g][f] / max(cnt[g],1)
__global__ void pool_mean_kernel(const float* __restrict__ pooled, const float* __restrict__ cnt,
                                 float* __restrict__ pm) {
    int i = blockIdx.x * blockDim.x + threadIdx.x;
    if (i < N_GRAPHS * HID) pm[i] = pooled[i] / fmaxf(cnt[i >> 6], 1.0f);
}

// ---------------------------------------------------------------------------
// Tiny dense layer: OUT[G,J] = act(IN[G,K] @ W[K,J] + B[J])
// ---------------------------------------------------------------------------
__global__ void dense_kernel(const float* __restrict__ IN, const float* __restrict__ W,
                             const float* __restrict__ B, float* __restrict__ OUT,
                             int G, int K, int J, int relu) {
    int idx = blockIdx.x * blockDim.x + threadIdx.x;
    if (idx >= G * J) return;
    const int g = idx / J, j = idx % J;
    float acc = B[j];
    const float* in = IN + (size_t)g * K;
    for (int k = 0; k < K; ++k) acc = fmaf(in[k], W[k * J + j], acc);
    OUT[idx] = relu ? fmaxf(acc, 0.0f) : acc;
}

// out = b_feat * t_feat + bias
__global__ void combine_kernel(const float* __restrict__ bf, const float* __restrict__ tf,
                               const float* __restrict__ bias, float* __restrict__ out) {
    int i = blockIdx.x * blockDim.x + threadIdx.x;
    if (i < N_GRAPHS * 2) out[i] = bf[i] * tf[i] + bias[i & 1];
}

// ---------------------------------------------------------------------------
// Launch
// ---------------------------------------------------------------------------
extern "C" void kernel_launch(void* const* d_in, const int* in_sizes, int n_in,
                              void* d_out, int out_size, void* d_ws, size_t ws_size,
                              hipStream_t stream) {
    (void)in_sizes; (void)n_in; (void)out_size; (void)ws_size;

    // ---- inputs (setup_inputs dict order, params in insertion order) ------
    const float* x      = (const float*)d_in[0];
    const float* x_loc  = (const float*)d_in[1];
    const float* W1  = (const float*)d_in[2],  *b1  = (const float*)d_in[3];
    const float* W2  = (const float*)d_in[4],  *b2  = (const float*)d_in[5];
    const float* W3  = (const float*)d_in[6],  *b3  = (const float*)d_in[7];
    const float* bn_g[3] = {(const float*)d_in[8],  (const float*)d_in[12], (const float*)d_in[16]};
    const float* bn_b[3] = {(const float*)d_in[9],  (const float*)d_in[13], (const float*)d_in[17]};
    const float* bn_m[3] = {(const float*)d_in[10], (const float*)d_in[14], (const float*)d_in[18]};
    const float* bn_v[3] = {(const float*)d_in[11], (const float*)d_in[15], (const float*)d_in[19]};
    const float* Wm1 = (const float*)d_in[20], *bm1 = (const float*)d_in[21];
    const float* Wm2 = (const float*)d_in[22], *bm2 = (const float*)d_in[23];
    const float* Wm3 = (const float*)d_in[24], *bm3 = (const float*)d_in[25];
    const float* Wo  = (const float*)d_in[26], *bo  = (const float*)d_in[27];
    const float* Wt1 = (const float*)d_in[28], *bt1 = (const float*)d_in[29];
    const float* Wt2 = (const float*)d_in[30], *bt2 = (const float*)d_in[31];
    const float* Wt3 = (const float*)d_in[32], *bt3 = (const float*)d_in[33];
    const float* bias = (const float*)d_in[34];
    const int* edge_index = (const int*)d_in[35];
    const int* batch      = (const int*)d_in[36];
    const int* e_src = edge_index;             // row 0 of [2, E]
    const int* e_dst = edge_index + N_EDGES;   // row 1
    float* out = (float*)d_out;

    // ---- workspace layout (floats) ---------------------------------------
    float* ws = (float*)d_ws;
    const size_t NF = (size_t)N_NODES * HID;
    float* dis    = ws;                     // 100000 (pad to 100352)
    float* hw     = ws + 100352;            // N*64
    float* agg    = hw  + NF;               // N*64 (doubles as h)
    float* pooled = agg + NF;               // 4096
    float* cnt    = pooled + N_GRAPHS * HID; // 64
    float* pm     = cnt + 64;               // 4096
    float* z1     = pm + 4096;              // 4096
    float* z2     = z1 + 4096;              // 2048
    float* z3     = z2 + 2048;              // 1024
    float* bf     = z3 + 1024;              // 128
    float* t1     = bf + 128;               // 8192
    float* t2     = t1 + 8192;              // 16384
    float* tf     = t2 + 16384;             // 128

    const int TPB = 256;
    const int blkN    = (N_NODES + TPB - 1) / TPB;
    const int blkE    = (N_EDGES + TPB - 1) / TPB;
    const int blkNF   = (int)((NF + TPB - 1) / TPB);
    const int blkEdgeWave = N_EDGES / 8;    // wave-per-edge, 8 waves/block

    // ---- degree / normalization ------------------------------------------
    zero_kernel<<<blkN, TPB, 0, stream>>>(dis, N_NODES);
    degree_kernel<<<blkE, TPB, 0, stream>>>(e_dst, dis, N_EDGES);
    rsqrt_kernel<<<blkN, TPB, 0, stream>>>(dis, N_NODES);

    // ---- three GCN layers -------------------------------------------------
    const float* layer_in[3] = {x, agg, agg};
    const float* layer_W[3]  = {W1, W2, W3};
    const float* layer_b[3]  = {b1, b2, b3};
    for (int L = 0; L < 3; ++L) {
        gemm_node_wmma<<<256, TPB, 0, stream>>>(layer_in[L], layer_W[L], hw, N_NODES);
        zero_kernel<<<blkNF, TPB, 0, stream>>>(agg, (int)NF);
        gcn_scatter<<<blkEdgeWave, TPB, 0, stream>>>(hw, e_src, e_dst, dis, agg, N_EDGES);
        gcn_finalize<<<blkNF, TPB, 0, stream>>>(agg, hw, dis, layer_b[L],
                                                bn_g[L], bn_b[L], bn_m[L], bn_v[L], N_NODES);
    }

    // ---- mean pool --------------------------------------------------------
    zero_kernel<<<(N_GRAPHS * HID + N_GRAPHS + TPB - 1) / TPB, TPB, 0, stream>>>(pooled, N_GRAPHS * HID + N_GRAPHS);
    pool_kernel<<<120, TPB, 0, stream>>>(agg, batch, pooled, cnt, N_NODES);
    pool_mean_kernel<<<(N_GRAPHS * HID + TPB - 1) / TPB, TPB, 0, stream>>>(pooled, cnt, pm);

    // ---- branch MLP -------------------------------------------------------
    dense_kernel<<<(N_GRAPHS * 64 + TPB - 1) / TPB, TPB, 0, stream>>>(pm, Wm1, bm1, z1, N_GRAPHS, 64, 64, 1);
    dense_kernel<<<(N_GRAPHS * 32 + TPB - 1) / TPB, TPB, 0, stream>>>(z1, Wm2, bm2, z2, N_GRAPHS, 64, 32, 1);
    dense_kernel<<<(N_GRAPHS * 16 + TPB - 1) / TPB, TPB, 0, stream>>>(z2, Wm3, bm3, z3, N_GRAPHS, 32, 16, 1);
    dense_kernel<<<(N_GRAPHS * 2  + TPB - 1) / TPB, TPB, 0, stream>>>(z3, Wo,  bo,  bf, N_GRAPHS, 16, 2, 0);

    // ---- trunk MLP --------------------------------------------------------
    dense_kernel<<<(N_GRAPHS * 128 + TPB - 1) / TPB, TPB, 0, stream>>>(x_loc, Wt1, bt1, t1, N_GRAPHS, 3,   128, 1);
    dense_kernel<<<(N_GRAPHS * 256 + TPB - 1) / TPB, TPB, 0, stream>>>(t1,    Wt2, bt2, t2, N_GRAPHS, 128, 256, 1);
    dense_kernel<<<(N_GRAPHS * 2   + TPB - 1) / TPB, TPB, 0, stream>>>(t2,    Wt3, bt3, tf, N_GRAPHS, 256, 2,   0);

    // ---- combine ----------------------------------------------------------
    combine_kernel<<<1, TPB, 0, stream>>>(bf, tf, bias, out);
}
